// PersistentMemoryReader_74406013436034
// MI455X (gfx1250) — matrix operand order repaired and checked by reading, not compiled, not run on hardware
//
#include <hip/hip_runtime.h>
#include <math.h>

// Problem dims (fixed by the reference): B=4, L=2048 -> M = 8192 rows; D=512; S=8192.
#define MTOT 8192
#define DDIM 512
#define STOT 8192
#define EPSF 1e-8f
#define NEGINF (-1e9f)

typedef __attribute__((ext_vector_type(16))) _Float16 v16h;
typedef __attribute__((ext_vector_type(8)))  float    v8f;

static __device__ __forceinline__ v8f wmma16(v16h a, v16h b, v8f c) {
  // D = A(16x32 f16) * B(32x16 f16) + C(16x16 f32)
  return __builtin_amdgcn_wmma_f32_16x16x32_f16(false, a, false, b, (short)0, c,
                                                false, false);
}

// 32 contiguous bytes (16 halves) -> v16h  (B-matrix fragment: K-contiguous)
static __device__ __forceinline__ v16h ld32B(const _Float16* p) {
  v16h r;
  ((uint4*)&r)[0] = ((const uint4*)p)[0];
  ((uint4*)&r)[1] = ((const uint4*)p)[1];
  return r;
}
// two 16-byte chunks (A-matrix fragment: adjacent chunks, K-contiguous)
static __device__ __forceinline__ v16h ld16x2(const _Float16* p0, const _Float16* p1) {
  v16h r;
  ((uint4*)&r)[0] = *(const uint4*)p0;
  ((uint4*)&r)[1] = *(const uint4*)p1;
  return r;
}

// ---------------------------------------------------------------------------
// Kernel 1: convert K/V to f16 planes, V transposed [D,S]; kmag; qmagsq init.
// grid = S blocks, 256 threads
// ---------------------------------------------------------------------------
__global__ void pmr_prep(const float* __restrict__ keys,
                         const float* __restrict__ values,
                         _Float16* __restrict__ k_rh, _Float16* __restrict__ k_ih,
                         _Float16* __restrict__ v_rt, _Float16* __restrict__ v_it,
                         float* __restrict__ kmag, float* __restrict__ qmagsq) {
  const int s = blockIdx.x;
  const int t = threadIdx.x;
  __shared__ float red[256];
  float acc = 0.f;
  const float2* k2 = (const float2*)keys;
  const float2* v2 = (const float2*)values;
  for (int d = t; d < DDIM; d += 256) {
    float2 kv = k2[s * DDIM + d];
    k_rh[s * DDIM + d] = (_Float16)kv.x;
    k_ih[s * DDIM + d] = (_Float16)kv.y;
    acc += kv.x * kv.x + kv.y * kv.y;
    float2 vv = v2[s * DDIM + d];
    v_rt[d * STOT + s] = (_Float16)vv.x;   // transposed for contiguous PV B-frags
    v_it[d * STOT + s] = (_Float16)vv.y;
  }
  red[t] = acc;
  __syncthreads();
  for (int o = 128; o > 0; o >>= 1) {
    if (t < o) red[t] += red[t + o];
    __syncthreads();
  }
  if (t == 0) {
    kmag[s] = sqrtf(red[0] + EPSF);
    qmagsq[s] = EPSF;   // S == M here; pre-seed EPS for atomic accumulation
  }
}

// ---------------------------------------------------------------------------
// Kernel 2: complex Q projection  q = x @ (W_r + iW_i)^T  via WMMA f16.
// grid = (N/128=4, M/16=512), 256 threads (8 waves, one 16x16 N-tile each)
// ---------------------------------------------------------------------------
__global__ void pmr_qgemm(const float* __restrict__ x,
                          const float* __restrict__ W_r,
                          const float* __restrict__ W_i,
                          _Float16* __restrict__ q_rh, _Float16* __restrict__ q_ih,
                          float* __restrict__ qmagsq) {
  const int wv  = threadIdx.x >> 5;
  const int l   = threadIdx.x & 31;
  const int col = l & 15;      // A row / B col / C col
  const int hf  = l >> 4;
  const int m0  = blockIdx.y * 16;
  const int n0  = blockIdx.x * 128 + wv * 16;

  v8f accr = {};
  v8f acci = {};
  const float2* x2 = (const float2*)x;   // interleaved (re,im)
  const int mrow = m0 + col;
  const int ecol = n0 + col;

  for (int kb = 0; kb < DDIM; kb += 32) {
    v16h ar, ai, br, bi, bin;
    _Float16* arp  = (_Float16*)&ar;
    _Float16* aip  = (_Float16*)&ai;
    _Float16* brp  = (_Float16*)&br;
    _Float16* bip  = (_Float16*)&bi;
    _Float16* binp = (_Float16*)&bin;
#pragma unroll
    for (int j = 0; j < 8; j++) {
      float2 f = x2[mrow * DDIM + kb + hf * 8 + j];
      arp[j] = (_Float16)f.x;  aip[j] = (_Float16)f.y;
      float2 g = x2[mrow * DDIM + kb + 16 + hf * 8 + j];
      arp[j + 8] = (_Float16)g.x;  aip[j + 8] = (_Float16)g.y;
    }
#pragma unroll
    for (int j = 0; j < 16; j++) {
      int d = kb + (j < 8 ? hf * 8 + j : 16 + hf * 8 + (j - 8));
      float wr = W_r[ecol * DDIM + d];
      float wi = W_i[ecol * DDIM + d];
      brp[j]  = (_Float16)wr;
      bip[j]  = (_Float16)wi;
      binp[j] = (_Float16)(-wi);
    }
    accr = wmma16(ar, br,  accr);   // + x_r W_r
    accr = wmma16(ai, bin, accr);   // - x_i W_i  (negated B fragment; f16 WMMA has no A/B NEG)
    acci = wmma16(ar, bi,  acci);   // + x_r W_i
    acci = wmma16(ai, br,  acci);   // + x_i W_r
  }

#pragma unroll
  for (int v = 0; v < 8; v++) {
    int m = m0 + v + 8 * hf;
    int n = n0 + col;
    q_rh[m * DDIM + n] = (_Float16)accr[v];
    q_ih[m * DDIM + n] = (_Float16)acci[v];
    float p = accr[v] * accr[v] + acci[v] * acci[v];
    p += __shfl_xor(p, 1, 32);
    p += __shfl_xor(p, 2, 32);
    p += __shfl_xor(p, 4, 32);
    p += __shfl_xor(p, 8, 32);
    if (col == 0) atomicAdd(&qmagsq[m], p);   // row sum over this tile's 16 cols
  }
}

// ---------------------------------------------------------------------------
// Kernel 3: fused scores -> online softmax -> PV -> complex RMS norm.
// grid = M/16 = 512 blocks, 256 threads (8 waves).
// Block owns 16 query rows (Q staged once in LDS); loops S in chunks of 256
// (32 keys per wave). Each wave owns 64 output columns for PV.
// ---------------------------------------------------------------------------
__global__ void pmr_attn(const _Float16* __restrict__ q_rh,
                         const _Float16* __restrict__ q_ih,
                         const _Float16* __restrict__ k_rh,
                         const _Float16* __restrict__ k_ih,
                         const _Float16* __restrict__ v_rt,
                         const _Float16* __restrict__ v_it,
                         const float* __restrict__ kmag,
                         const float* __restrict__ qmagsq,
                         const int* __restrict__ mask,
                         const float* __restrict__ gamma,
                         float* __restrict__ out) {
  __shared__ __align__(16) _Float16 qs_r[16 * DDIM];   // 16 KB: Q real plane
  __shared__ __align__(16) _Float16 qs_i[16 * DDIM];   // 16 KB: Q imag plane
  __shared__ __align__(16) float    sc[16 * 256];      // 16 KB: chunk scores
  __shared__ __align__(16) _Float16 pbuf[16 * 256];    //  8 KB: chunk probs (f16)
  __shared__ float red[16 * 16];
  __shared__ float run_max[16], run_sum[16], factor[16], mrow[16];
  __shared__ float qmg[16], rowsum[16], rsinv[16], rminv[16];

  const int tid = threadIdx.x;
  const int wv  = tid >> 5;
  const int l   = tid & 31;
  const int col = l & 15;
  const int hf  = l >> 4;
  const int m0  = blockIdx.x * 16;

  // Stage this block's 16 Q rows (contiguous in memory) into LDS once.
  {
    const uint4* gqr = (const uint4*)(q_rh + (size_t)m0 * DDIM);
    const uint4* gqi = (const uint4*)(q_ih + (size_t)m0 * DDIM);
    uint4* sqr = (uint4*)qs_r;
    uint4* sqi = (uint4*)qs_i;
    for (int i = tid; i < 16 * DDIM / 8; i += 256) {
      sqr[i] = gqr[i];
      sqi[i] = gqi[i];
    }
  }
  if (tid < 16) {
    run_max[tid] = -3.0e38f;
    run_sum[tid] = 0.f;
    rowsum[tid]  = 0.f;
    qmg[tid]     = sqrtf(qmagsq[m0 + tid]);   // already includes +EPS seed
  }
  __syncthreads();

  v8f or_[4], oi_[4];
#pragma unroll
  for (int t = 0; t < 4; t++) { or_[t] = (v8f){}; oi_[t] = (v8f){}; }

  for (int c = 0; c < STOT / 256; c++) {
    const int sbase = c * 256;
    const int s0w   = sbase + wv * 32;

    // ---- scores: dot(q,k) over D=512 (two 16x16 S-tiles per wave) ----
    v8f dt[2];
    dt[0] = (v8f){}; dt[1] = (v8f){};
    for (int kb = 0; kb < DDIM; kb += 32) {
      v16h aqr = ld16x2(qs_r + col * DDIM + kb + hf * 16,
                        qs_r + col * DDIM + kb + hf * 16 + 8);
      v16h aqi = ld16x2(qs_i + col * DDIM + kb + hf * 16,
                        qs_i + col * DDIM + kb + hf * 16 + 8);
#pragma unroll
      for (int t = 0; t < 2; t++) {
        int s = s0w + t * 16 + col;
        v16h bkr = ld32B(k_rh + s * DDIM + kb + hf * 16);
        dt[t] = wmma16(aqr, bkr, dt[t]);
        v16h bki = ld32B(k_ih + s * DDIM + kb + hf * 16);
        dt[t] = wmma16(aqi, bki, dt[t]);
      }
    }
    // scale by 1/(|q||k|+eps), mask, spill to LDS
#pragma unroll
    for (int t = 0; t < 2; t++) {
      int s  = s0w + t * 16 + col;
      float km = kmag[s];
      int   mk = mask[s];
      int   lc = wv * 32 + t * 16 + col;
#pragma unroll
      for (int v = 0; v < 8; v++) {
        int r = v + 8 * hf;
        float sv = mk ? (dt[t][v] / (qmg[r] * km + EPSF)) : NEGINF;
        sc[r * 256 + lc] = sv;
      }
    }
    __syncthreads();

    // ---- online softmax over this 256-key chunk ----
    {
      int r = tid >> 4, j = tid & 15;
      float lm = -3.0e38f;
#pragma unroll 4
      for (int tt = 0; tt < 16; tt++) lm = fmaxf(lm, sc[r * 256 + j * 16 + tt]);
      red[r * 16 + j] = lm;
    }
    __syncthreads();
    if (tid < 16) {
      float cm = red[tid * 16];
      for (int j = 1; j < 16; j++) cm = fmaxf(cm, red[tid * 16 + j]);
      float mn = fmaxf(run_max[tid], cm);
      factor[tid] = __expf(run_max[tid] - mn);
      mrow[tid]   = mn;
      run_max[tid] = mn;
    }
    __syncthreads();
    {
      int r = tid >> 4, j = tid & 15;
      float mn = mrow[r], ls = 0.f;
#pragma unroll 4
      for (int tt = 0; tt < 16; tt++) {
        int idx = r * 256 + j * 16 + tt;
        float p = __expf(sc[idx] - mn);
        pbuf[idx] = (_Float16)p;
        ls += p;
      }
      red[r * 16 + j] = ls;
    }
    __syncthreads();
    if (tid < 16) {
      float s2 = 0.f;
      for (int j = 0; j < 16; j++) s2 += red[tid * 16 + j];
      run_sum[tid] = run_sum[tid] * factor[tid] + s2;
    }
    __syncthreads();

    // ---- rescale running accumulators, then PV over the chunk ----
    float fv[8];
#pragma unroll
    for (int v = 0; v < 8; v++) fv[v] = factor[v + 8 * hf];
#pragma unroll
    for (int t = 0; t < 4; t++)
#pragma unroll
      for (int v = 0; v < 8; v++) { or_[t][v] *= fv[v]; oi_[t][v] *= fv[v]; }

    for (int kb = 0; kb < 256; kb += 32) {
      v16h ap = ld16x2(&pbuf[col * 256 + kb + hf * 16],
                       &pbuf[col * 256 + kb + hf * 16 + 8]);
      int sb = sbase + kb + hf * 16;
#pragma unroll
      for (int t = 0; t < 4; t++) {
        int d = wv * 64 + t * 16 + col;
        v16h bvr = ld32B(v_rt + d * STOT + sb);
        or_[t] = wmma16(ap, bvr, or_[t]);
        v16h bvi = ld32B(v_it + d * STOT + sb);
        oi_[t] = wmma16(ap, bvi, oi_[t]);
      }
    }
    __syncthreads();
  }

  // ---- epilogue: /sum, complex RMS norm, *gamma, store interleaved ----
  if (tid < 16) rsinv[tid] = 1.0f / (run_sum[tid] + 1e-20f);
  __syncthreads();

  float rp[8];
#pragma unroll
  for (int v = 0; v < 8; v++) rp[v] = 0.f;
#pragma unroll
  for (int t = 0; t < 4; t++)
#pragma unroll
    for (int v = 0; v < 8; v++) {
      float inv = rsinv[v + 8 * hf];
      float a = or_[t][v] * inv;
      float b = oi_[t][v] * inv;
      or_[t][v] = a; oi_[t][v] = b;
      rp[v] += a * a + b * b;
    }
#pragma unroll
  for (int v = 0; v < 8; v++) {
    float p = rp[v];
    p += __shfl_xor(p, 1, 32);
    p += __shfl_xor(p, 2, 32);
    p += __shfl_xor(p, 4, 32);
    p += __shfl_xor(p, 8, 32);
    if (col == 0) atomicAdd(&rowsum[v + 8 * hf], p);
  }
  __syncthreads();
  if (tid < 16) rminv[tid] = rsqrtf(rowsum[tid] / (float)DDIM + EPSF);
  __syncthreads();

  float2* out2 = (float2*)out;   // [..., D, 2] -> float2 per (m,d)
#pragma unroll
  for (int t = 0; t < 4; t++) {
    int d = wv * 64 + t * 16 + col;
    float g = gamma[d];
#pragma unroll
    for (int v = 0; v < 8; v++) {
      int m = m0 + v + 8 * hf;
      float s2 = rminv[v + 8 * hf] * g;
      float2 o;
      o.x = or_[t][v] * s2;
      o.y = oi_[t][v] * s2;
      out2[m * DDIM + d] = o;
    }
  }
}

// ---------------------------------------------------------------------------
extern "C" void kernel_launch(void* const* d_in, const int* in_sizes, int n_in,
                              void* d_out, int out_size, void* d_ws, size_t ws_size,
                              hipStream_t stream) {
  const float* x      = (const float*)d_in[0];   // [4,2048,512,2]
  const float* keys   = (const float*)d_in[1];   // [8192,512,2]
  const float* values = (const float*)d_in[2];   // [8192,512,2]
  const int*   mask   = (const int*)d_in[3];     // [8192]
  const float* W_r    = (const float*)d_in[4];   // [512,512]
  const float* W_i    = (const float*)d_in[5];   // [512,512]
  const float* gamma  = (const float*)d_in[6];   // [512]
  float* out = (float*)d_out;

  // workspace carve: 6 f16 planes of 4M elements (8 MB each) + 2 f32 vectors
  _Float16* k_rh = (_Float16*)d_ws;
  _Float16* k_ih = k_rh + (size_t)STOT * DDIM;
  _Float16* v_rt = k_ih + (size_t)STOT * DDIM;
  _Float16* v_it = v_rt + (size_t)STOT * DDIM;
  _Float16* q_rh = v_it + (size_t)STOT * DDIM;
  _Float16* q_ih = q_rh + (size_t)MTOT * DDIM;
  float* kmag    = (float*)(q_ih + (size_t)MTOT * DDIM);
  float* qmagsq  = kmag + STOT;

  pmr_prep<<<STOT, 256, 0, stream>>>(keys, values, k_rh, k_ih, v_rt, v_it,
                                     kmag, qmagsq);
  pmr_qgemm<<<dim3(DDIM / 128, MTOT / 16), 256, 0, stream>>>(
      x, W_r, W_i, q_rh, q_ih, qmagsq);
  pmr_attn<<<MTOT / 16, 256, 0, stream>>>(q_rh, q_ih, k_rh, k_ih, v_rt, v_it,
                                          kmag, qmagsq, mask, gamma, out);
}
